// ZImageAttention_21895743275334
// MI455X (gfx1250) — compile-verified, matrix operand order checked by reading, not run on hardware
//
#include <hip/hip_runtime.h>

typedef __attribute__((ext_vector_type(16))) __bf16 v16bf;
typedef __attribute__((ext_vector_type(8)))  __bf16 v8bf;
typedef __attribute__((ext_vector_type(8)))  float  v8f;
typedef __attribute__((ext_vector_type(4)))  int    v4i;
typedef unsigned short bfraw;

// address-space qualified int4 for the async global->LDS builtin
typedef __attribute__((address_space(1))) v4i gv4i;   // global
typedef __attribute__((address_space(3))) v4i lv4i;   // LDS

#define DIM      3072
#define NHEADS   24
#define HD       128
#define SEQ      2048
#define QKV_OUT  9216
#define EPS_RMS  1e-6f

__device__ __forceinline__ bfraw f2bfraw(float f) {
  return __builtin_bit_cast(bfraw, (__bf16)f);
}

// Build a 16x bf16 WMMA fragment from two contiguous 8-element (16B) chunks
// at p and p+16 (both 16B aligned by construction) -> 2x b128 loads.
__device__ __forceinline__ v16bf make_frag(const bfraw* p) {
  v8bf lo = *(const v8bf*)(p);
  v8bf hi = *(const v8bf*)(p + 16);
  return __builtin_shufflevector(lo, hi, 0, 1, 2, 3, 4, 5, 6, 7,
                                 8, 9, 10, 11, 12, 13, 14, 15);
}

// ---------------------------------------------------------------------------
// gfx1250 async global->LDS copy (16B per lane). Guarded so the file compiles
// on toolchains where the builtin is absent (falls back to VGPR round-trip).
// ---------------------------------------------------------------------------
#if __has_builtin(__builtin_amdgcn_global_load_async_to_lds_b128)
#define ASYNC_LDS 1
#else
#define ASYNC_LDS 0
#endif

__device__ __forceinline__ void async_copy16(void* lds, const void* gsrc) {
#if ASYNC_LDS
  __builtin_amdgcn_global_load_async_to_lds_b128((gv4i*)gsrc, (lv4i*)lds, 0, 0);
#else
  *(uint4*)lds = *(const uint4*)gsrc;
#endif
}

__device__ __forceinline__ void wait_async() {
#if ASYNC_LDS
#if __has_builtin(__builtin_amdgcn_s_wait_asynccnt)
  __builtin_amdgcn_s_wait_asynccnt(0);
#else
  asm volatile("s_wait_asynccnt 0" ::: "memory");
#endif
#endif
}

// ---------------------------------------------------------------------------
// Bulk fp32 -> bf16 conversion (one-time, so GEMM inner loops are pure
// load+wmma). n must be a multiple of 4.
// ---------------------------------------------------------------------------
__global__ __launch_bounds__(256)
void cvt_f32_bf16(const float* __restrict__ in, bfraw* __restrict__ out, int n) {
  int i = (int)(blockIdx.x * blockDim.x + threadIdx.x) * 4;
  if (i + 3 >= n) return;
  float4 f = *(const float4*)(in + i);
  ushort4 r;
  r.x = f2bfraw(f.x);
  r.y = f2bfraw(f.y);
  r.z = f2bfraw(f.z);
  r.w = f2bfraw(f.w);
  *(ushort4*)(out + i) = r;
}

// ---------------------------------------------------------------------------
// C[M,N] = A[M,K] * B[N,K]^T   (A,B bf16 row-major, fp32 accum/output)
// block = 128 threads (4 waves); wave -> 16(M) x 64(N); block -> 64 x 64 tile
// ---------------------------------------------------------------------------
__global__ __launch_bounds__(128)
void gemm_nt_bf16(const bfraw* __restrict__ A,
                  const bfraw* __restrict__ B,
                  float* __restrict__ C,
                  int M, int N, int K) {
  const int lane = threadIdx.x & 31;
  const int wave = threadIdx.x >> 5;
  const int half = lane >> 4;     // 0/1: which K-half group of lanes
  const int lrow = lane & 15;     // row (A) / col (B,C) within tile
  const int m0 = blockIdx.y * 64 + wave * 16;
  const int n0 = blockIdx.x * 64;

  const bfraw* arow = A + (size_t)(m0 + lrow) * K + 8 * half;
  const bfraw* brow = B + (size_t)(n0 + lrow) * K + 8 * half;

  v8f acc[4] = {};

  for (int k0 = 0; k0 < K; k0 += 32) {
    v16bf a = make_frag(arow + k0);
#pragma unroll
    for (int t = 0; t < 4; ++t) {
      v16bf b = make_frag(brow + (size_t)t * 16 * K + k0);
      acc[t] = __builtin_amdgcn_wmma_f32_16x16x32_bf16(
          false, a, false, b, (short)0, acc[t], false, false);
    }
  }

  // C/D layout: element i -> row m0 + i + 8*half, col n0 + t*16 + lrow
#pragma unroll
  for (int t = 0; t < 4; ++t)
#pragma unroll
    for (int i = 0; i < 8; ++i)
      C[(size_t)(m0 + i + 8 * half) * N + n0 + t * 16 + lrow] = acc[t][i];
}

// ---------------------------------------------------------------------------
// Per-(s,head) RMS-norm on q,k + interleaved RoPE + bf16 pack of q,k,v into
// [head][s][d] layout. One wave32 per (s,head); 4 elements/lane.
// ---------------------------------------------------------------------------
__global__ __launch_bounds__(128)
void qk_norm_rope(const float* __restrict__ qkv,
                  const float* __restrict__ cosb,
                  const float* __restrict__ sinb,
                  const float* __restrict__ wq,
                  const float* __restrict__ wk,
                  bfraw* __restrict__ qb,
                  bfraw* __restrict__ kb,
                  bfraw* __restrict__ vb) {
  const int gwave = (int)((blockIdx.x * blockDim.x + threadIdx.x) >> 5);
  const int lane  = threadIdx.x & 31;
  const int s = gwave / NHEADS;
  const int h = gwave % NHEADS;
  if (s >= SEQ) return;

  const float* qp = qkv + (size_t)s * QKV_OUT + h * HD;
  const float* kp = qp + NHEADS * HD;
  const float* vp = kp + NHEADS * HD;
  const int e0 = lane * 4;

  float q[4], k[4], v[4];
  float sq = 0.f, sk = 0.f;
#pragma unroll
  for (int j = 0; j < 4; ++j) {
    q[j] = qp[e0 + j]; k[j] = kp[e0 + j]; v[j] = vp[e0 + j];
    sq += q[j] * q[j]; sk += k[j] * k[j];
  }
#pragma unroll
  for (int off = 16; off > 0; off >>= 1) {
    sq += __shfl_xor(sq, off, 32);
    sk += __shfl_xor(sk, off, 32);
  }
  const float invq = rsqrtf(sq * (1.0f / HD) + EPS_RMS);
  const float invk = rsqrtf(sk * (1.0f / HD) + EPS_RMS);
#pragma unroll
  for (int j = 0; j < 4; ++j) {
    q[j] = q[j] * invq * wq[e0 + j];
    k[j] = k[j] * invk * wk[e0 + j];
  }

  const size_t ob = ((size_t)h * SEQ + s) * HD + e0;
#pragma unroll
  for (int pr = 0; pr < 2; ++pr) {
    const int p = (e0 >> 1) + pr;
    const float c  = cosb[(size_t)s * (HD / 2) + p];
    const float sn = sinb[(size_t)s * (HD / 2) + p];
    const float q1 = q[2 * pr], q2 = q[2 * pr + 1];
    const float k1 = k[2 * pr], k2 = k[2 * pr + 1];
    qb[ob + 2 * pr]     = f2bfraw(q1 * c - q2 * sn);
    qb[ob + 2 * pr + 1] = f2bfraw(q2 * c + q1 * sn);
    kb[ob + 2 * pr]     = f2bfraw(k1 * c - k2 * sn);
    kb[ob + 2 * pr + 1] = f2bfraw(k2 * c + k1 * sn);
  }
#pragma unroll
  for (int j = 0; j < 4; ++j) vb[ob + j] = f2bfraw(v[j]);
}

// ---------------------------------------------------------------------------
// Streaming-softmax attention, one head per blockIdx.x, 128 q-rows per block
// (8 waves x 16 rows). K tile staged via async global->LDS B128 DMA
// (ASYNCcnt); V tile transposed through VGPRs (overlaps the async copies).
// P restaged via per-wave LDS region for C-layout -> A-layout conversion.
// ---------------------------------------------------------------------------
__global__ __launch_bounds__(256)
void flash_attn(const bfraw* __restrict__ qb,
                const bfraw* __restrict__ kb,
                const bfraw* __restrict__ vb,
                const unsigned char* __restrict__ mask,
                bfraw* __restrict__ ao) {
  __shared__ __align__(16) bfraw ktile[32 * HD];      // [key][d]
  __shared__ __align__(16) bfraw vtile[HD * 32];      // [d][key] (transposed)
  __shared__ __align__(16) bfraw ptile[8][16 * 32];   // per-wave P [qrow][key]

  const int head = blockIdx.x;
  const int lane = threadIdx.x & 31;
  const int wave = threadIdx.x >> 5;
  const int half = lane >> 4;
  const int lrow = lane & 15;
  const int q0 = blockIdx.y * 128 + wave * 16;

  // Q A-fragments for all 4 K-steps of d (resident for whole kernel)
  v16bf aq[4];
  {
    const bfraw* qrow = qb + ((size_t)head * SEQ + q0 + lrow) * HD + 8 * half;
#pragma unroll
    for (int kk = 0; kk < 4; ++kk) aq[kk] = make_frag(qrow + kk * 32);
  }

  float mrow[8], lsum[8];
  v8f o[8] = {};
#pragma unroll
  for (int i = 0; i < 8; ++i) { mrow[i] = -3.0e38f; lsum[i] = 0.f; }

  const float scale = 0.08838834764831843f;  // 1/sqrt(128)

  for (int kb0 = 0; kb0 < SEQ; kb0 += 32) {
    __syncthreads();
    // K tile: async DMA straight into LDS, 16B (8 bf16) per chunk
    for (int c = threadIdx.x; c < 32 * (HD / 8); c += 256) {
      const int key = c >> 4;            // 16 chunks per 128-elem row
      const int dc  = (c & 15) * 8;
      async_copy16(&ktile[key * HD + dc],
                   kb + ((size_t)head * SEQ + kb0 + key) * HD + dc);
    }
    // V tile: transpose into [d][key] via VGPRs while K-tile DMA is in flight
    for (int idx = threadIdx.x; idx < 32 * HD; idx += 256) {
      const int key = idx >> 7;
      const int d   = idx & (HD - 1);
      vtile[d * 32 + key] = vb[((size_t)head * SEQ + kb0 + key) * HD + d];
    }
    wait_async();
    __syncthreads();

    // S = Q K^T for 32 keys (two 16-key WMMA tiles)
    v8f s0 = {}, s1 = {};
#pragma unroll
    for (int kk = 0; kk < 4; ++kk) {
      v16bf b0 = make_frag(&ktile[lrow * HD + kk * 32 + 8 * half]);
      v16bf b1 = make_frag(&ktile[(16 + lrow) * HD + kk * 32 + 8 * half]);
      s0 = __builtin_amdgcn_wmma_f32_16x16x32_bf16(false, aq[kk], false, b0, (short)0, s0, false, false);
      s1 = __builtin_amdgcn_wmma_f32_16x16x32_bf16(false, aq[kk], false, b1, (short)0, s1, false, false);
    }

    const bool mk0 = mask[kb0 + lrow] != 0;
    const bool mk1 = mask[kb0 + 16 + lrow] != 0;
#pragma unroll
    for (int i = 0; i < 8; ++i) {
      s0[i] = mk0 ? s0[i] * scale : -3.0e38f;
      s1[i] = mk1 ? s1[i] * scale : -3.0e38f;
    }

    // online softmax: per C-element i (q-row = i + 8*half), reduce across the
    // 16-lane half-wave that holds that row's key columns
    float p0[8], p1[8], corr[8];
#pragma unroll
    for (int i = 0; i < 8; ++i) {
      float vmax = fmaxf(s0[i], s1[i]);
#pragma unroll
      for (int off = 8; off > 0; off >>= 1) vmax = fmaxf(vmax, __shfl_xor(vmax, off, 32));
      const float mn = fmaxf(mrow[i], vmax);
      corr[i] = __expf(mrow[i] - mn);
      mrow[i] = mn;
      p0[i] = __expf(s0[i] - mn);
      p1[i] = __expf(s1[i] - mn);
      float r = p0[i] + p1[i];
#pragma unroll
      for (int off = 8; off > 0; off >>= 1) r += __shfl_xor(r, off, 32);
      lsum[i] = lsum[i] * corr[i] + r;
    }
#pragma unroll
    for (int dt = 0; dt < 8; ++dt)
#pragma unroll
      for (int i = 0; i < 8; ++i) o[dt][i] *= corr[i];

    // P: C-layout -> row-major LDS -> A-fragment (wave-private, DS in-order)
#pragma unroll
    for (int i = 0; i < 8; ++i) {
      ptile[wave][(i + 8 * half) * 32 + lrow]      = f2bfraw(p0[i]);
      ptile[wave][(i + 8 * half) * 32 + 16 + lrow] = f2bfraw(p1[i]);
    }
    v16bf pa = make_frag(&ptile[wave][lrow * 32 + 8 * half]);

    // O += P V  (8 d-tiles of 16 columns)
#pragma unroll
    for (int dt = 0; dt < 8; ++dt) {
      v16bf bv = make_frag(&vtile[(dt * 16 + lrow) * 32 + 8 * half]);
      o[dt] = __builtin_amdgcn_wmma_f32_16x16x32_bf16(false, pa, false, bv, (short)0, o[dt], false, false);
    }
  }

  // normalize and write [s][h*128+d] bf16 for the output projection
#pragma unroll
  for (int dt = 0; dt < 8; ++dt)
#pragma unroll
    for (int i = 0; i < 8; ++i)
      ao[(size_t)(q0 + i + 8 * half) * DIM + head * HD + dt * 16 + lrow] =
          f2bfraw(o[dt][i] / lsum[i]);
}

// ---------------------------------------------------------------------------
extern "C" void kernel_launch(void* const* d_in, const int* in_sizes, int n_in,
                              void* d_out, int out_size, void* d_ws, size_t ws_size,
                              hipStream_t stream) {
  (void)in_sizes; (void)n_in; (void)out_size; (void)ws_size;
  const float*         hidden = (const float*)d_in[0];
  const unsigned char* msk    = (const unsigned char*)d_in[1];
  const float*         cosb   = (const float*)d_in[2];
  const float*         sinb   = (const float*)d_in[3];
  const float*         w_qkv  = (const float*)d_in[4];
  const float*         wq     = (const float*)d_in[5];
  const float*         wk     = (const float*)d_in[6];
  const float*         w_out  = (const float*)d_in[7];
  float*               out    = (float*)d_out;

  char* ws = (char*)d_ws;
  size_t off = 0;
  float* qkv    = (float*)(ws + off); off += (size_t)SEQ * QKV_OUT * sizeof(float);
  bfraw* qb     = (bfraw*)(ws + off); off += (size_t)NHEADS * SEQ * HD * sizeof(bfraw);
  bfraw* kbuf   = (bfraw*)(ws + off); off += (size_t)NHEADS * SEQ * HD * sizeof(bfraw);
  bfraw* vbuf   = (bfraw*)(ws + off); off += (size_t)NHEADS * SEQ * HD * sizeof(bfraw);
  bfraw* hbf    = (bfraw*)(ws + off); off += (size_t)SEQ * DIM * sizeof(bfraw);
  bfraw* wqkvbf = (bfraw*)(ws + off); off += (size_t)QKV_OUT * DIM * sizeof(bfraw);
  bfraw* woutbf = (bfraw*)(ws + off); off += (size_t)DIM * DIM * sizeof(bfraw);
  bfraw* aobf   = (bfraw*)(ws + off); off += (size_t)SEQ * DIM * sizeof(bfraw);

  // 0) one-time fp32 -> bf16 conversions (4 elems/thread)
  {
    int n1 = SEQ * DIM;        // hidden
    int n2 = QKV_OUT * DIM;    // w_qkv
    int n3 = DIM * DIM;        // w_out
    cvt_f32_bf16<<<n1 / 1024, 256, 0, stream>>>(hidden, hbf, n1);
    cvt_f32_bf16<<<n2 / 1024, 256, 0, stream>>>(w_qkv, wqkvbf, n2);
    cvt_f32_bf16<<<n3 / 1024, 256, 0, stream>>>(w_out, woutbf, n3);
  }

  // 1) QKV projection: [2048,9216] = hidden[2048,3072] @ w_qkv[9216,3072]^T
  dim3 g1(QKV_OUT / 64, SEQ / 64);
  gemm_nt_bf16<<<g1, 128, 0, stream>>>(hbf, wqkvbf, qkv, SEQ, QKV_OUT, DIM);

  // 2) per-head RMS-norm + RoPE + bf16 pack
  const int nwaves = SEQ * NHEADS;          // 49152 waves, 4 per block
  qk_norm_rope<<<nwaves / 4, 128, 0, stream>>>(qkv, cosb, sinb, wq, wk, qb, kbuf, vbuf);

  // 3) attention -> bf16 [s][dim]
  dim3 g3(NHEADS, SEQ / 128);
  flash_attn<<<g3, 256, 0, stream>>>(qb, kbuf, vbuf, msk, aobf);

  // 4) output projection: [2048,3072] = ao @ w_out[3072,3072]^T
  dim3 g4(DIM / 64, SEQ / 64);
  gemm_nt_bf16<<<g4, 128, 0, stream>>>(aobf, woutbf, out, SEQ, DIM, DIM);
}
